// ModulatedConv2d_72378788873040
// MI455X (gfx1250) — matrix operand order repaired and checked by reading, not compiled, not run on hardware
//
#include <hip/hip_runtime.h>

// ModulatedConv2d (StyleGAN2) for MI455X / gfx1250: wave32, WMMA f16->f32,
// TDM (tensor_load_to_lds) double-buffered weight slabs, LDS-staged x tiles.
// B=16, Cin=Cout=128, K=3, H=W=128, style=512.

#define BATCH 16
#define CIN   128
#define COUT  128
#define KS    3
#define HH    128
#define WW    128
#define SDIM  512
#define KDIM  (CIN * KS * KS)   // 1152

#define NW      64              // output pixels per block (one row segment)
#define TCOLS   (NW + 2)        // 66, halo columns
#define CSTRIDE 144             // padded cin stride in LDS halfs (288B = 9*32)

// A-slab in LDS: 128 cout rows x 128 cin halfs (256B) + 16B pad -> 272B rows
#define AROW_H   136            // 272 bytes / 2
#define ASLAB_B  (COUT * AROW_H * 2)          // 34816 bytes per buffer
#define XTILE_OFF (2 * ASLAB_B)               // 69632
#define XTILE_B  (KS * TCOLS * CSTRIDE * 2)   // 57024
#define SMEM_BYTES (XTILE_OFF + XTILE_B)      // 126656

typedef __attribute__((ext_vector_type(16))) _Float16 v16h;
typedef __attribute__((ext_vector_type(8)))  _Float16 v8h;
typedef __attribute__((ext_vector_type(2)))  _Float16 v2h;
typedef __attribute__((ext_vector_type(8)))  float    v8f;
typedef __attribute__((ext_vector_type(4)))  unsigned int u32x4;
typedef __attribute__((ext_vector_type(8)))  unsigned int u32x8;

__device__ __forceinline__ void wait_tensorcnt0() {
#if defined(__has_builtin) && __has_builtin(__builtin_amdgcn_s_wait_tensorcnt)
    __builtin_amdgcn_s_wait_tensorcnt(0);
#else
    asm volatile("s_wait_tensorcnt 0x0" ::: "memory");
#endif
}

// Issue one TDM load: 2D tile (tile_dim0 = 128 cin halfs, tile_dim1 = 128 cout
// rows, row stride 1152 halfs in global) -> contiguous LDS with 16B pad after
// every 64 dwords (one row), giving a 272B LDS row stride.
// D# fields per CDNA5 ISA.
__device__ __forceinline__ void tdm_load_a_slab(const _Float16* gptr,
                                                unsigned lds_byte_off) {
    unsigned long long ga = (unsigned long long)(uintptr_t)gptr;
    u32x4 g0;
    g0[0] = 1u;                                   // count=1, user descriptor
    g0[1] = lds_byte_off;                         // lds_addr
    g0[2] = (unsigned)(ga & 0xffffffffu);         // global_addr[31:0]
    g0[3] = (unsigned)((ga >> 32) & 0x01ffffffu)  // global_addr[56:32]
            | (2u << 30);                         // type = 2 ("image")
    u32x8 g1;
    g1[0] = (1u << 16)      // data_size = 1 -> 2 bytes
          | (1u << 20)      // pad_enable
          | (5u << 22)      // pad_interval: every 64 dwords (one 256B row)
          | (3u << 25);     // pad_amount: 4 dwords (16B)
    g1[1] = (128u << 16);   // tensor_dim0 = 128 (lo16 at bits[79:64-32])
    g1[2] = (128u << 16);   // tensor_dim0 hi = 0 | tensor_dim1 lo16 = 128
    g1[3] = (128u << 16);   // tensor_dim1 hi = 0 | tile_dim0 = 128
    g1[4] = 128u;           // tile_dim1 = 128 | tile_dim2 = 0
    g1[5] = (unsigned)KDIM; // tensor_dim0_stride lo32 = 1152
    g1[6] = 0u;             // stride0 hi16 | tensor_dim1_stride lo16 (unused)
    g1[7] = 0u;
    u32x4 gz = {0u, 0u, 0u, 0u};
    asm volatile("tensor_load_to_lds %0, %1, %2, %3"
                 :: "s"(g0), "s"(g1), "s"(gz), "s"(gz)
                 : "memory");
}

// ---------------------------------------------------------------------------
// Kernel 1: style affine + demod + pack per-sample modulated weights to f16.
// A layout: wf16[b][cout][kpos*128 + cin], kpos = kh*3 + kw.
// ---------------------------------------------------------------------------
__global__ void __launch_bounds__(128) modulate_pack_kernel(
    const float* __restrict__ style, const float* __restrict__ weight,
    const float* __restrict__ mod_w, const float* __restrict__ mod_b,
    _Float16* __restrict__ wf16)
{
    const int b = blockIdx.x;
    const int t = threadIdx.x;     // cin in phase 1, cout in phase 2
    __shared__ float s_sh[CIN];

    const float lin_scale = 0.044194173824159216f;   // 1/sqrt(512)
    {
        float acc = 0.f;
        const float* st = style + (size_t)b * SDIM;
        const float* mw = mod_w + (size_t)t * SDIM;
        for (int k = 0; k < SDIM; ++k) acc += st[k] * mw[k];
        s_sh[t] = acc * lin_scale + mod_b[t];
    }
    __syncthreads();

    const float scale = 0.029462782549439483f;       // 1/sqrt(128*9)
    const float* wrow = weight + (size_t)t * CIN * KS * KS;
    float ssq = 0.f;
    for (int cin = 0; cin < CIN; ++cin) {
        const float sv = s_sh[cin] * scale;
        #pragma unroll
        for (int kk = 0; kk < KS * KS; ++kk) {
            const float v = wrow[cin * (KS * KS) + kk] * sv;
            ssq += v * v;
        }
    }
    const float demod = rsqrtf(ssq + 1e-8f);

    _Float16* dst = wf16 + (size_t)(b * COUT + t) * KDIM;
    for (int cin = 0; cin < CIN; ++cin) {
        const float sv = s_sh[cin] * scale * demod;
        #pragma unroll
        for (int kk = 0; kk < KS * KS; ++kk)
            dst[kk * CIN + cin] = (_Float16)(wrow[cin * (KS * KS) + kk] * sv);
    }
}

// ---------------------------------------------------------------------------
// Kernel 2: implicit-GEMM conv. Block = 128 couts x 64 pixels of one sample.
// 8 wave32; each wave: 16 couts x 64 pixels = four 16x16 f32 tiles,
// 9 K-slabs (TDM double-buffered) x 4 cin-chunks x 4 n-tiles WMMAs.
// ---------------------------------------------------------------------------
__global__ void __launch_bounds__(256) modconv_wmma_kernel(
    const float* __restrict__ x, const _Float16* __restrict__ wf16,
    float* __restrict__ out)
{
    extern __shared__ char smem[];
    _Float16* aslab = (_Float16*)smem;                  // 2 x 34816 B
    _Float16* xtile = (_Float16*)(smem + XTILE_OFF);    // 57024 B

    const int wseg = blockIdx.x & 1;            // W/NW = 2 segments
    const int h    = (blockIdx.x >> 1) & 127;   // output row
    const int b    = blockIdx.x >> 8;           // sample
    const int w0   = wseg * NW;
    const int tid  = threadIdx.x;
    const int lane = tid & 31;
    const int wave = tid >> 5;

    const _Float16* Ag = wf16 + (size_t)b * COUT * KDIM;

    // Kick off TDM for K-slab 0 while everyone stages the x tile.
    if (wave == 0)
        tdm_load_a_slab(Ag, 0u);

    // ---- stage x halo tile (3 rows x 66 cols x 128 cin) as f16 into LDS ----
    // One cin-pair per step -> packed b32 LDS stores, w-coalesced global loads.
    const int total2 = KS * TCOLS * (CIN / 2);  // 12672
    for (int e = tid; e < total2; e += 256) {
        const int cin = (e / (KS * TCOLS)) * 2;
        const int rc  = e - (cin >> 1) * (KS * TCOLS);
        const int r   = rc / TCOLS;
        const int c   = rc - r * TCOLS;
        const int gy  = h - 1 + r;
        const int gx  = w0 - 1 + c;
        float v0 = 0.f, v1 = 0.f;
        if ((unsigned)gy < HH && (unsigned)gx < WW) {
            const size_t gbase = (((size_t)b * CIN + cin) * HH + gy) * WW + gx;
            v0 = x[gbase];
            v1 = x[gbase + (size_t)HH * WW];
        }
        v2h p; p[0] = (_Float16)v0; p[1] = (_Float16)v1;
        *(v2h*)&xtile[(r * TCOLS + c) * CSTRIDE + cin] = p;
    }

    const int m0   = wave * 16;        // cout base for this wave
    const int half = lane >> 4;        // 0: lanes 0-15, 1: lanes 16-31
    const int ln   = lane & 15;

    v8f acc0 = {}, acc1 = {}, acc2 = {}, acc3 = {};

    for (int kpos = 0; kpos < 9; ++kpos) {
        if (wave == 0)
            wait_tensorcnt0();          // slab kpos resident in LDS
        __syncthreads();                // publish slab (and x tile on kpos==0)
        if (wave == 0 && kpos < 8)      // prefetch next slab into other buffer
            tdm_load_a_slab(Ag + (size_t)(kpos + 1) * CIN,
                            (unsigned)(((kpos + 1) & 1) * ASLAB_B));

        const _Float16* slab = aslab + (kpos & 1) * (ASLAB_B / 2);
        const int kh = kpos / 3;
        const int kw = kpos - kh * 3;

        #pragma unroll
        for (int cc = 0; cc < 4; ++cc) {
            // A fragment (contiguous k order, same as B): lanes 0-15 hold K 0-15,
            // lanes 16-31 hold K 16-31; padded 272B LDS rows.
            const _Float16* ap = slab + (m0 + ln) * AROW_H + cc * 32 + half * 16;
            const v8h alo = *(const v8h*)(ap);
            const v8h ahi = *(const v8h*)(ap + 8);
            const v16h a = __builtin_shufflevector(alo, ahi,
                0, 1, 2, 3, 4, 5, 6, 7, 8, 9, 10, 11, 12, 13, 14, 15);

            // B fragments (32x16): lane = col n, 16 contiguous K halfs (32B).
            const _Float16* bp =
                &xtile[(kh * TCOLS + ln + kw) * CSTRIDE + cc * 32 + half * 16];
            const v16h b0 = *(const v16h*)(bp);
            const v16h b1 = *(const v16h*)(bp + 16 * CSTRIDE);
            const v16h b2 = *(const v16h*)(bp + 32 * CSTRIDE);
            const v16h b3 = *(const v16h*)(bp + 48 * CSTRIDE);

            acc0 = __builtin_amdgcn_wmma_f32_16x16x32_f16(
                false, a, false, b0, (short)0, acc0, false, false);
            acc1 = __builtin_amdgcn_wmma_f32_16x16x32_f16(
                false, a, false, b1, (short)0, acc1, false, false);
            acc2 = __builtin_amdgcn_wmma_f32_16x16x32_f16(
                false, a, false, b2, (short)0, acc2, false, false);
            acc3 = __builtin_amdgcn_wmma_f32_16x16x32_f16(
                false, a, false, b3, (short)0, acc3, false, false);
        }
    }

    // ---- epilogue: C/D layout -> VGPR j holds (M = j + 8*half, N = ln) ----
    float* obase = out + (((size_t)b * COUT + m0) * HH + h) * WW + w0;
    #pragma unroll
    for (int j = 0; j < 8; ++j) {
        const size_t row = (size_t)(j + half * 8) * HH * WW;
        obase[row + ln]      = acc0[j];
        obase[row + 16 + ln] = acc1[j];
        obase[row + 32 + ln] = acc2[j];
        obase[row + 48 + ln] = acc3[j];
    }
}

// ---------------------------------------------------------------------------
extern "C" void kernel_launch(void* const* d_in, const int* in_sizes, int n_in,
                              void* d_out, int out_size, void* d_ws, size_t ws_size,
                              hipStream_t stream) {
    const float* x      = (const float*)d_in[0];
    const float* style  = (const float*)d_in[1];
    const float* weight = (const float*)d_in[2];
    const float* mod_w  = (const float*)d_in[3];
    const float* mod_b  = (const float*)d_in[4];
    float* out          = (float*)d_out;

    // Workspace: packed per-sample modulated f16 weights [B][COUT][1152] = 4.5 MB
    _Float16* wf16 = (_Float16*)d_ws;

    modulate_pack_kernel<<<dim3(BATCH), dim3(128), 0, stream>>>(
        style, weight, mod_w, mod_b, wf16);

    modconv_wmma_kernel<<<dim3(BATCH * HH * (WW / NW)), dim3(256), SMEM_BYTES,
                          stream>>>(x, wf16, out);
}
